// MoE_48584670053012
// MI455X (gfx1250) — matrix operand order repaired and checked by reading, not compiled, not run on hardware
//
#include <hip/hip_runtime.h>
#include <hip/hip_bf16.h>

#define NEXP 8
#define DMODEL 512
#define ROWDW 520   // 512 f32 + 2x4-DWORD TDM pads (one per 256-DW interval)

typedef __attribute__((ext_vector_type(16))) __bf16   bf16x16;
typedef __attribute__((ext_vector_type(8)))  float    floatx8;
typedef __attribute__((ext_vector_type(4)))  float    floatx4;
typedef __attribute__((ext_vector_type(4)))  unsigned uintx4;
typedef __attribute__((ext_vector_type(8)))  int      intx8;
typedef __attribute__((ext_vector_type(4)))  int      intx4;

// ---------------------------------------------------------------------------
// ws layout (bytes):
//   0    : float importance[8]
//   32   : float load[8]
//   64   : int   tk_idx[N*2]
//   64+8N: float tk_gate[N*2]
//   next : __bf16 wbt[8][512][512]   (weights transposed to [e][n][k], bf16)
// ---------------------------------------------------------------------------

__global__ void moe_zero_stats(float* stats) {
    if (threadIdx.x < 16) stats[threadIdx.x] = 0.0f;
}

// wbt[e][n][k] = bf16( w[e][k][n] )
__global__ void moe_wtrans(const float* __restrict__ w, __bf16* __restrict__ wbt) {
    int idx = blockIdx.x * blockDim.x + threadIdx.x;      // e*2^18 + n*2^9 + k
    int k = idx & 511;
    int n = (idx >> 9) & 511;
    int e = idx >> 18;
    wbt[idx] = (__bf16)w[(e << 18) + (k << 9) + n];
}

// One wave per token: logits = x_row @ w_gate, top-2 + softmax, stats atomics.
__global__ void moe_gate(const float* __restrict__ x, const float* __restrict__ wg,
                         int* __restrict__ tk_idx, float* __restrict__ tk_gate,
                         float* __restrict__ stats) {
    const int lane = threadIdx.x & 31;
    const int wave = threadIdx.x >> 5;
    const int t = blockIdx.x * 8 + wave;

    const float* xr = x + (size_t)t * DMODEL + lane * 16;
    float lg[NEXP];
#pragma unroll
    for (int e = 0; e < NEXP; ++e) lg[e] = 0.0f;

#pragma unroll
    for (int j = 0; j < 16; ++j) {
        float xv = xr[j];
        const float* wr = wg + (size_t)(lane * 16 + j) * NEXP;
#pragma unroll
        for (int e = 0; e < NEXP; ++e) lg[e] += xv * wr[e];
    }
#pragma unroll
    for (int e = 0; e < NEXP; ++e) {
#pragma unroll
        for (int off = 16; off >= 1; off >>= 1)
            lg[e] += __shfl_xor(lg[e], off, 32);
    }

    if (lane == 0) {
        float v0 = -1e30f, v1 = -1e30f;
        int i0 = 0, i1 = 0;
#pragma unroll
        for (int e = 0; e < NEXP; ++e) {
            float v = lg[e];
            if (v > v0) { v1 = v0; i1 = i0; v0 = v; i0 = e; }
            else if (v > v1) { v1 = v; i1 = e; }
        }
        float e1 = __expf(v1 - v0);
        float inv = 1.0f / (1.0f + e1);
        float g0 = inv, g1 = e1 * inv;
        tk_idx[2 * t]      = i0;
        tk_idx[2 * t + 1]  = i1;
        tk_gate[2 * t]     = g0;
        tk_gate[2 * t + 1] = g1;
        atomicAdd(&stats[i0], g0);
        atomicAdd(&stats[i1], g1);
        atomicAdd(&stats[8 + i0], 1.0f);
        atomicAdd(&stats[8 + i1], 1.0f);
    }
}

// padded LDS dword offset for element (row r, k) of the f32 x tile
__device__ __forceinline__ int pad_dw(int r, int k) {
    return r * ROWDW + k + ((k >= 256) ? 4 : 0);
}

// Main kernel: block = 16 tokens x 512 cols. 8 waves, each wave owns 4 N-tiles.
// x tile staged to LDS (f32, TDM-padded layout) by the Tensor Data Mover;
// A fragments built once per wave in registers (bf16), reused over 8 experts
// and 4 N-tiles.  y_tile = sum_e diag(g_e) * (X @ W_e) with f32 WMMA accum.
__global__ void __launch_bounds__(256)
moe_main(const float* __restrict__ x, const __bf16* __restrict__ wbt,
         const int* __restrict__ tk_idx, const float* __restrict__ tk_gate,
         float* __restrict__ y) {
    __shared__ __align__(16) float shx[16 * ROWDW];   // 33,280 B
    __shared__ float shg[NEXP][16];
    __shared__ unsigned shmask;

    const int tid = threadIdx.x;
    const int m0  = blockIdx.x * 16;

    if (tid == 0) shmask = 0u;
    __syncthreads();

    // Dense per-tile gate table + expert presence mask
    if (tid < 128) {
        int e = tid >> 4, r = tid & 15;
        int t = m0 + r;
        int e0 = tk_idx[2 * t], e1 = tk_idx[2 * t + 1];
        float g = 0.0f;
        if (e0 == e) g += tk_gate[2 * t];
        if (e1 == e) g += tk_gate[2 * t + 1];
        shg[e][r] = g;
        if (e == 0) atomicOr(&shmask, (1u << e0) | (1u << e1));
    }

#if __has_builtin(__builtin_amdgcn_tensor_load_to_lds)
    // ---- Tensor Data Mover: DMA the 16x512 f32 tile into padded LDS ----
    if (tid == 0) {
        unsigned lds_base = (unsigned)(size_t)(void*)&shx[0];
        unsigned long long ga =
            (unsigned long long)(const void*)(x + ((size_t)m0 << 9));
        uintx4 g0;
        g0.x = 1u;                                   // count=1 (valid user D#)
        g0.y = lds_base;                             // lds_addr
        g0.z = (unsigned)(ga & 0xffffffffu);         // global_addr[31:0]
        g0.w = (unsigned)((ga >> 32) & 0x1ffffffu)   // global_addr[56:32]
               | (2u << 30);                         // type = 2 (image)
        intx8 g1;
        g1[0] = (int)((2u << 16)      // data_size = 4B
                      | (1u << 20)    // pad_enable
                      | (7u << 22)    // pad_interval: every 256 DWORDs
                      | (3u << 25));  // pad_amount: 4 DWORDs
        g1[1] = (int)(512u << 16);    // tensor_dim0 = 512 (bits 63:48)
        g1[2] = 0;                    // tensor_dim0 hi / tensor_dim1 lo16
        g1[3] = (int)(1u | (512u << 16)); // tensor_dim1 = 65536; tile_dim0 = 512
        g1[4] = 16;                   // tile_dim1 = 16 rows; tile_dim2 = 0
        g1[5] = 512;                  // tensor_dim0_stride = 512 elements
        g1[6] = 0;
        g1[7] = 0;
        intx4 z4 = {0, 0, 0, 0};
#if defined(__clang_major__) && __clang_major__ >= 23
        intx8 z8 = {0, 0, 0, 0, 0, 0, 0, 0};
        __builtin_amdgcn_tensor_load_to_lds(g0, g1, z4, z4, z8, 0);
#else
        __builtin_amdgcn_tensor_load_to_lds(g0, g1, z4, z4, 0);
#endif
#if __has_builtin(__builtin_amdgcn_s_wait_tensorcnt)
        __builtin_amdgcn_s_wait_tensorcnt(0);
#else
        asm volatile("s_wait_tensorcnt 0x0" ::: "memory");
#endif
    }
#else
    // ---- Fallback: manual cooperative staging into the same padded layout ----
#pragma unroll
    for (int j = 0; j < 8; ++j) {
        int i4 = tid + j * 256;              // float4 granule, 128 per row
        int r  = i4 >> 7;
        int c4 = (i4 & 127) * 4;
        floatx4 v = *(const floatx4*)(x + ((size_t)(m0 + r) << 9) + c4);
        *(floatx4*)(shx + pad_dw(r, c4)) = v;
    }
#endif
    __syncthreads();

    const int lane = tid & 31;
    const int wave = tid >> 5;
    const int lh   = lane >> 4;          // lane-half (K split per ISA A/B layout)
    const int ll   = lane & 15;
    const int n0   = wave * 64;          // 4 N-tiles of 16 cols per wave
    const unsigned emask = shmask;

    // Build all 16 A fragments (K=0..511) once, in registers (bf16).
    // A layout: lanes 0-15 row=ll K {k0..k0+15};
    //           lanes 16-31 same row set, K offset +16 (lh*16).
    bf16x16 Afr[16];
#pragma unroll
    for (int kk = 0; kk < 16; ++kk) {
        int k  = kk * 32 + lh * 16;
        floatx8 lo = *(const floatx8*)(shx + pad_dw(ll, k));
        floatx8 hi = *(const floatx8*)(shx + pad_dw(ll, k + 8));
        bf16x16 a;
#pragma unroll
        for (int j = 0; j < 8; ++j) { a[j] = (__bf16)lo[j]; a[j + 8] = (__bf16)hi[j]; }
        Afr[kk] = a;
    }

    // B fragment base: column n0+ll, K base lh*16 (contiguous k = 16*lh + s, same order as A)
    const __bf16* wbB = wbt + ((size_t)(n0 + ll) << 9) + (lh << 4);

    floatx8 out[4];
#pragma unroll
    for (int nt = 0; nt < 4; ++nt)
        out[nt] = floatx8{0.f, 0.f, 0.f, 0.f, 0.f, 0.f, 0.f, 0.f};

    for (int e = 0; e < NEXP; ++e) {
        if (!((emask >> e) & 1u)) continue;
        const __bf16* wb = wbB + ((size_t)e << 18);
        __builtin_prefetch(wb, 0, 0);
        floatx8 acc[4];
#pragma unroll
        for (int nt = 0; nt < 4; ++nt)
            acc[nt] = floatx8{0.f, 0.f, 0.f, 0.f, 0.f, 0.f, 0.f, 0.f};
#pragma unroll
        for (int kk = 0; kk < 16; ++kk) {
#pragma unroll
            for (int nt = 0; nt < 4; ++nt) {
                bf16x16 B = *(const bf16x16*)(wb + ((size_t)(nt * 16) << 9) + kk * 32);
                acc[nt] = __builtin_amdgcn_wmma_f32_16x16x32_bf16(
                    false, Afr[kk], false, B, (short)0, acc[nt], false, false);
            }
        }
        // per-row gate: C/D row = i + 8*lane_half
#pragma unroll
        for (int nt = 0; nt < 4; ++nt)
#pragma unroll
            for (int i = 0; i < 8; ++i)
                out[nt][i] += shg[e][i + (lh << 3)] * acc[nt][i];
    }

    // Store: lane covers col n0 + nt*16 + ll, rows m0 + 8*lh + i
#pragma unroll
    for (int nt = 0; nt < 4; ++nt) {
        float* yr = y + ((size_t)(m0 + (lh << 3)) << 9) + n0 + nt * 16 + ll;
#pragma unroll
        for (int i = 0; i < 8; ++i) yr[(size_t)i << 9] = out[nt][i];
    }
}

__global__ void moe_loss(const float* __restrict__ stats, float* __restrict__ out_loss) {
    if (threadIdx.x == 0) {
        float loss = 0.0f;
#pragma unroll
        for (int s = 0; s < 2; ++s) {
            const float* v = stats + s * 8;
            float mean = 0.0f;
#pragma unroll
            for (int e = 0; e < NEXP; ++e) mean += v[e];
            mean *= (1.0f / 8.0f);
            float var = 0.0f;
#pragma unroll
            for (int e = 0; e < NEXP; ++e) {
                float d = v[e] - mean;
                var += d * d;
            }
            var *= (1.0f / 7.0f);                       // ddof = 1
            loss += var / (mean * mean + 1e-10f);
        }
        *out_loss = loss * 0.01f;
    }
}

extern "C" void kernel_launch(void* const* d_in, const int* in_sizes, int n_in,
                              void* d_out, int out_size, void* d_ws, size_t ws_size,
                              hipStream_t stream) {
    const float* x  = (const float*)d_in[0];   // (8, 8192, 512) f32
    const float* wg = (const float*)d_in[1];   // (512, 8) f32
    const float* we = (const float*)d_in[2];   // (8, 512, 512) f32

    const int N = in_sizes[0] / DMODEL;        // 65536 tokens

    char* ws = (char*)d_ws;
    float*  stats   = (float*)ws;                                   // 16 floats
    int*    tk_idx  = (int*)(ws + 64);                              // N*2 ints
    float*  tk_gate = (float*)(ws + 64 + (size_t)N * 2 * 4);        // N*2 floats
    __bf16* wbt     = (__bf16*)(ws + 64 + (size_t)N * 4 * 4);       // 8*512*512 bf16

    float* y        = (float*)d_out;                                // N*512 f32
    float* out_loss = (float*)d_out + (out_size - 1);               // scalar

    moe_zero_stats<<<1, 32, 0, stream>>>(stats);
    moe_wtrans<<<(NEXP * DMODEL * DMODEL) / 256, 256, 0, stream>>>(we, wbt);
    moe_gate<<<N / 8, 256, 0, stream>>>(x, wg, tk_idx, tk_gate, stats);

    moe_main<<<dim3(N / 16, 1), 256, 0, stream>>>(x, wbt, tk_idx, tk_gate, y);

    moe_loss<<<1, 32, 0, stream>>>(stats, out_loss);
}
